// GAT_54142357733608
// MI455X (gfx1250) — matrix operand order repaired and checked by reading, not compiled, not run on hardware
//
#include <hip/hip_runtime.h>
#include <hip/hip_fp16.h>

// GAT forward: 3 layers, N=10000 nodes, E=160000 edges, C=256 ch, H=8 heads.
// GEMMs via v_wmma_f32_16x16x32_f16 (f16 in, f32 accum); edge phase via
// hardware global_atomic_add_f32 (bandwidth-bound: ~4GB traffic / launch).

typedef __attribute__((ext_vector_type(8)))  _Float16 v8h;
typedef __attribute__((ext_vector_type(16))) _Float16 v16h;
typedef __attribute__((ext_vector_type(8)))  float    v8f;

#define NODES 10000
#define EDGES 160000
#define CH    256
#define MPAD  10016          // 626 * 16
#define GEMM_WAVES 8
#define GEMM_NT    4         // 4 x 16 = 64 output cols per wave

static __device__ __forceinline__ float warp_sum32(float v) {
  #pragma unroll
  for (int off = 16; off > 0; off >>= 1) v += __shfl_xor(v, off, 32);
  return v;
}

static __device__ __forceinline__ void atomAddF32(float* a, float v) {
  unsafeAtomicAdd(a, v);     // -> global_atomic_add_f32 (no CAS loop)
}

// ---------------- utility: zero fill ----------------
__global__ void k_zero(float* __restrict__ p, int n) {
  int i = blockIdx.x * blockDim.x + threadIdx.x;
  if (i < n) p[i] = 0.f;
}

// ---------------- fp32 [rows,K] -> f16 [MPAD,K], zero-pad rows ----------------
__global__ void k_cvt_a(const float* __restrict__ in, _Float16* __restrict__ out,
                        int rows, int K) {
  int i = blockIdx.x * blockDim.x + threadIdx.x;
  if (i >= MPAD * K) return;
  int r = i / K;
  out[i] = (r < rows) ? (_Float16)in[i] : (_Float16)0.f;
}

// ---------------- W [K,Nout] fp32 -> Wt [Nout,K] f16 (transpose) ----------------
__global__ void k_cvt_wt(const float* __restrict__ W, _Float16* __restrict__ Wt,
                         int K, int Nout) {
  int i = blockIdx.x * blockDim.x + threadIdx.x;
  if (i >= K * Nout) return;
  int n = i / K, k = i - n * K;
  Wt[i] = (_Float16)W[(size_t)k * Nout + n];
}

// ---------------- GEMM: C[MPAD,Nout] = A[MPAD,K] * Wt[Nout,K]^T ----------------
// One wave computes a 16x64 strip of C via 4 accumulators; K-step 32.
__global__ void k_gemm_wmma(const _Float16* __restrict__ A,
                            const _Float16* __restrict__ Bt,
                            float* __restrict__ C,
                            int K, int Nout) {
  const int lane = threadIdx.x & 31;
  const int wave = threadIdx.x >> 5;
  const int rowTile = blockIdx.x * GEMM_WAVES + wave;
  if (rowTile * 16 >= MPAD) return;            // wave-uniform exit
  const int colBase = blockIdx.y * (GEMM_NT * 16);
  const int l15 = lane & 15;
  const int kg  = lane >> 4;                   // 0 or 1: K-half group

  v8f acc[GEMM_NT] = {};

  const _Float16* pa = A + (size_t)(rowTile * 16 + l15) * K + kg * 8;
  for (int k = 0; k < K; k += 32) {
    // A-frag: halfs 0..7 = K(kg*8..+7), halfs 8..15 = K(kg*8+16..+23)
    v8h alo = *(const v8h*)(pa + k);
    v8h ahi = *(const v8h*)(pa + k + 16);
    v16h afrag = __builtin_shufflevector(alo, ahi,
        0,1,2,3,4,5,6,7,8,9,10,11,12,13,14,15);
    #pragma unroll
    for (int t = 0; t < GEMM_NT; ++t) {
      // B-frag: lane col = l15, halfs 0..7 = K(kg*8..+7), halfs 8..15 = K(kg*8+16..+23)
      const _Float16* pb = Bt + (size_t)(colBase + t * 16 + l15) * K + k + kg * 8;
      v8h blo = *(const v8h*)pb;
      v8h bhi = *(const v8h*)(pb + 16);
      v16h bfrag = __builtin_shufflevector(blo, bhi,
          0,1,2,3,4,5,6,7,8,9,10,11,12,13,14,15);
      acc[t] = __builtin_amdgcn_wmma_f32_16x16x32_f16(
          false, afrag, false, bfrag, (short)0, acc[t], false, false);
    }
  }
  // C layout: VGPR r -> row kg*8 + r, col l15
  const int crow = rowTile * 16 + kg * 8;
  #pragma unroll
  for (int t = 0; t < GEMM_NT; ++t) {
    float* pc = C + (size_t)crow * Nout + colBase + t * 16 + l15;
    #pragma unroll
    for (int r = 0; r < 8; ++r) pc[(size_t)r * Nout] = acc[t][r];
  }
}

// ---------------- attention logits: al_s/al_d[n,heads] = <h[n,hd,:], a[hd,:]> ----------------
__global__ void k_attn_logits(const float* __restrict__ h,
                              const float* __restrict__ a_src,
                              const float* __restrict__ a_dst,
                              float* __restrict__ al_s, float* __restrict__ al_d,
                              int heads) {
  int gw   = (int)((blockIdx.x * blockDim.x + threadIdx.x) >> 5);
  int lane = threadIdx.x & 31;
  if (gw >= NODES * heads) return;
  int node = gw / heads, hd = gw - node * heads;
  const float* hp = h + (size_t)node * heads * CH + (size_t)hd * CH;
  const float* as = a_src + (size_t)hd * CH;
  const float* ad = a_dst + (size_t)hd * CH;
  float ss = 0.f, sd = 0.f;
  #pragma unroll
  for (int j = 0; j < CH / 32; ++j) {
    float hv = hp[lane + j * 32];
    ss += hv * as[lane + j * 32];
    sd += hv * ad[lane + j * 32];
  }
  ss = warp_sum32(ss);
  sd = warp_sum32(sd);
  if (lane == 0) { al_s[gw] = ss; al_d[gw] = sd; }
}

// ---------------- edge pass 1: p = exp(leakyrelu(als[src]+ald[dst])), s[dst] += p ----------------
__global__ void k_edge_p(const int* __restrict__ src, const int* __restrict__ dst,
                         const float* __restrict__ al_s, const float* __restrict__ al_d,
                         float* __restrict__ p, float* __restrict__ s, int heads) {
  int i = blockIdx.x * blockDim.x + threadIdx.x;
  if (i >= EDGES * heads) return;
  int ed = i / heads, hd = i - ed * heads;
  float v = al_s[src[ed] * heads + hd] + al_d[dst[ed] * heads + hd];
  v = (v > 0.f) ? v : 0.2f * v;                 // leaky_relu(0.2)
  float pv = expf(v);                           // max-shift dropped (shift-invariant)
  p[i] = pv;
  atomAddF32(&s[dst[ed] * heads + hd], pv);
}

// ---------------- edge pass 2: agg[dst] += alpha * h[src] ----------------
__global__ void k_edge_agg(const int* __restrict__ src, const int* __restrict__ dst,
                           const float* __restrict__ h, const float* __restrict__ p,
                           const float* __restrict__ s, float* __restrict__ agg,
                           int heads) {
  int ed = blockIdx.x;
  int sN = src[ed], dN = dst[ed];
  int F = heads * CH;
  for (int f = threadIdx.x; f < F; f += blockDim.x) {
    int hd = f >> 8;                            // CH == 256
    float alpha = p[ed * heads + hd] / (s[dN * heads + hd] + 1e-16f);
    atomAddF32(&agg[(size_t)dN * F + f], h[(size_t)sN * F + f] * alpha);
  }
}

// ---------------- epilogue: out = elu(agg + b) ----------------
__global__ void k_epilogue(const float* __restrict__ agg, const float* __restrict__ b,
                           float* __restrict__ out, int F) {
  int i = blockIdx.x * blockDim.x + threadIdx.x;
  if (i >= NODES * F) return;
  int f = i % F;
  float v = agg[i] + b[f];
  out[i] = (v > 0.f) ? v : (expf(v) - 1.f);
}

extern "C" void kernel_launch(void* const* d_in, const int* in_sizes, int n_in,
                              void* d_out, int out_size, void* d_ws, size_t ws_size,
                              hipStream_t stream) {
  (void)in_sizes; (void)n_in; (void)out_size; (void)ws_size;
  const float* x   = (const float*)d_in[0];
  const int*   ei  = (const int*)d_in[1];
  const int* src = ei;
  const int* dst = ei + EDGES;
  const float* W1  = (const float*)d_in[2];
  const float* as1 = (const float*)d_in[3];
  const float* ad1 = (const float*)d_in[4];
  const float* b1  = (const float*)d_in[5];
  const float* W2  = (const float*)d_in[6];
  const float* as2 = (const float*)d_in[7];
  const float* ad2 = (const float*)d_in[8];
  const float* b2  = (const float*)d_in[9];
  const float* W3  = (const float*)d_in[10];
  const float* as3 = (const float*)d_in[11];
  const float* ad3 = (const float*)d_in[12];
  const float* b3  = (const float*)d_in[13];

  const int Fmax = 8 * CH;                      // 2048
  char* ws = (char*)d_ws;
  size_t off = 0;
  auto walloc = [&](size_t bytes) -> void* {
    void* pp = (void*)(ws + off);
    off += (bytes + 255) & ~(size_t)255;
    return pp;
  };
  _Float16* Af16 = (_Float16*)walloc((size_t)MPAD * Fmax * sizeof(_Float16));
  _Float16* Wt16 = (_Float16*)walloc((size_t)Fmax * Fmax * sizeof(_Float16));
  float* buf0 = (float*)walloc((size_t)MPAD * Fmax * sizeof(float));  // h (gemm out)
  float* buf1 = (float*)walloc((size_t)MPAD * Fmax * sizeof(float));
  float* buf2 = (float*)walloc((size_t)MPAD * Fmax * sizeof(float));
  float* al_s = (float*)walloc((size_t)NODES * 8 * sizeof(float));
  float* al_d = (float*)walloc((size_t)NODES * 8 * sizeof(float));
  float* pbuf = (float*)walloc((size_t)EDGES * 8 * sizeof(float));
  float* sbuf = (float*)walloc((size_t)NODES * 8 * sizeof(float));

  auto run_layer = [&](const float* in, int Kin, const float* W,
                       const float* asc, const float* adc, const float* bias,
                       int heads, float* hbuf, float* aggbuf, float* outbuf) {
    int F = heads * CH;
    int totA = MPAD * Kin;
    k_cvt_a<<<(totA + 255) / 256, 256, 0, stream>>>(in, Af16, NODES, Kin);
    int totW = Kin * F;
    k_cvt_wt<<<(totW + 255) / 256, 256, 0, stream>>>(W, Wt16, Kin, F);
    dim3 ggrid((MPAD / 16 + GEMM_WAVES - 1) / GEMM_WAVES, F / (GEMM_NT * 16));
    k_gemm_wmma<<<ggrid, GEMM_WAVES * 32, 0, stream>>>(Af16, Wt16, hbuf, Kin, F);
    int nwaves = NODES * heads;
    k_attn_logits<<<(nwaves * 32 + 255) / 256, 256, 0, stream>>>(
        hbuf, asc, adc, al_s, al_d, heads);
    k_zero<<<(NODES * heads + 255) / 256, 256, 0, stream>>>(sbuf, NODES * heads);
    k_zero<<<(NODES * F + 255) / 256, 256, 0, stream>>>(aggbuf, NODES * F);
    k_edge_p<<<(EDGES * heads + 255) / 256, 256, 0, stream>>>(
        src, dst, al_s, al_d, pbuf, sbuf, heads);
    k_edge_agg<<<EDGES, 256, 0, stream>>>(src, dst, hbuf, pbuf, sbuf, aggbuf, heads);
    k_epilogue<<<(NODES * F + 255) / 256, 256, 0, stream>>>(aggbuf, bias, outbuf, F);
  };

  // Layer 1: [N,64] -> [N,2048]
  run_layer(x,    64,   W1, as1, ad1, b1, 8, buf0, buf1, buf1);
  // Layer 2: [N,2048] -> [N,2048]
  run_layer(buf1, 2048, W2, as2, ad2, b2, 8, buf0, buf2, buf2);
  // Layer 3: [N,2048] -> [N,256]
  run_layer(buf2, 2048, W3, as3, ad3, b3, 1, buf0, buf1, (float*)d_out);
}
